// GeometricBasinCompatibility_75806172774558
// MI455X (gfx1250) — matrix operand order repaired and checked by reading, not compiled, not run on hardware
//
#include <hip/hip_runtime.h>

#define DI __device__ __forceinline__

typedef __attribute__((ext_vector_type(16))) __bf16 v16bf;
typedef __attribute__((ext_vector_type(8)))  float  v8f;

constexpr int Bn  = 4096;
constexpr int Cn  = 1000;
constexpr int Cp  = 1024;          // C padded to 64 tiles of 16
constexpr int Ln  = 32;
constexpr int Fn  = 64;
constexpr int Kn  = Ln * Fn;       // 2048 = folded GEMM K
constexpr int KS  = Kn / 32;       // 64 k-steps of 32
constexpr int KH  = 128;           // hier GEMM K (early|late · F)
constexpr int KSH = KH / 32;       // 4

// ---- workspace layout (bytes) ----
constexpr size_t SZ_A   = (size_t)Bn * Kn * 2;   // staged bf16 A
constexpr size_t SZ_B   = (size_t)Cp * Kn * 2;   // staged bf16 B
constexpr size_t SZ_AH  = (size_t)Bn * KH * 2;
constexpr size_t SZ_BH  = (size_t)Cp * KH * 2;
constexpr size_t OFF_AHI  = 0;
constexpr size_t OFF_ALO  = OFF_AHI  + SZ_A;
constexpr size_t OFF_BHI  = OFF_ALO  + SZ_A;
constexpr size_t OFF_BLO  = OFF_BHI  + SZ_B;
constexpr size_t OFF_AHHI = OFF_BLO  + SZ_B;
constexpr size_t OFF_AHLO = OFF_AHHI + SZ_AH;
constexpr size_t OFF_BHHI = OFF_AHLO + SZ_AH;
constexpr size_t OFF_BHLO = OFF_BHHI + SZ_BH;
constexpr size_t OFF_SST  = OFF_BHLO + SZ_BH;            // self_sim, [31][Bn] f32
constexpr size_t SZ_SST   = (size_t)31 * Bn * 4;
constexpr size_t OFF_EXT  = OFF_SST  + SZ_SST;           // expected, [31][Cp] f32

// ---- bf16 helpers (round-to-nearest-even, manual to avoid header deps) ----
DI unsigned short f2bf(float x) {
  union { float f; unsigned u; } v; v.f = x;
  unsigned r = v.u + 0x7FFFu + ((v.u >> 16) & 1u);
  return (unsigned short)(r >> 16);
}
DI float bf2f(unsigned short b) {
  union { float f; unsigned u; } v; v.u = ((unsigned)b) << 16;
  return v.f;
}

// ---- WMMA fragment element placement (contiguous k = 16h + e, same for A and B) ----
// 16-bit A matrix 16x32: lane = (h<<4)|m ; VGPR j, half 'bit' ; K = f(h,j,bit)
DI int stagedA_idx(int row, int kk, int ksteps) {
  int ks  = kk >> 5, k = kk & 31;
  int h   = (k >> 4) & 1;
  int j   = (k >> 1) & 7;
  int bit = k & 1;
  int lane = (h << 4) | (row & 15);
  int e    = (j << 1) | bit;
  return ((((row >> 4) * ksteps + ks) * 32 + lane) << 4) + e;
}
// 16-bit B 32x16 as staged here (contiguous k): lanes 0-15 K=0..15 (VGPR j: K=2j,2j+1), lanes 16-31 K=16..31
DI int stagedB_idx(int col, int kk, int ksteps) {
  int ks  = kk >> 5, k = kk & 31;
  int h   = (k >> 4) & 1;
  int j   = (k >> 1) & 7;
  int bit = k & 1;
  int lane = (h << 4) | (col & 15);
  int e    = (j << 1) | bit;
  return ((((col >> 4) * ksteps + ks) * 32 + lane) << 4) + e;
}

// WGP-scope streaming prefetch (scope field 0 = WGP: pull into all cache
// levels on miss, stop on WGP hit). __builtin_prefetch can only reach
// SE/SYS scope, which skips the near cache, so use the ISA op directly.
// Offset is a literal so all look-ahead slots share one base VGPR pair.
#define PF(p, offlit) \
  asm volatile("global_prefetch_b8 %0, off offset:" #offlit :: "v"(p))

// =====================  class-side preprocessing  =====================
__global__ void __launch_bounds__(256) prep_classes(
    const float* __restrict__ sig, const float* __restrict__ lres,
    unsigned short* __restrict__ Bhi, unsigned short* __restrict__ Blo,
    unsigned short* __restrict__ BHhi, unsigned short* __restrict__ BHlo,
    float* __restrict__ exT)
{
  int c = blockIdx.x;                 // 0..Cp-1 (>=Cn writes zeros)
  int t = threadIdx.x;                // 256 threads
  bool valid = (c < Cn);
  __shared__ float s_raw[Ln];
  __shared__ float s_p[Ln];
  __shared__ float s_vec[2 * Fn];
  __shared__ float s_ssq[2];

  if (t < Ln) s_raw[t] = valid ? lres[c * Ln + t] : 0.0f;
  __syncthreads();
  if (t < Ln) {                       // softmax over L (wave 0)
    float r = s_raw[t];
    float m = r;
    for (int o = 1; o < 32; o <<= 1) m = fmaxf(m, __shfl_xor(m, o, 32));
    float e = __expf(r - m);
    float s = e;
    for (int o = 1; o < 32; o <<= 1) s += __shfl_xor(s, o, 32);
    s_p[t] = e / s;
  }
  __syncthreads();

  // W[c,l,f] = 0.5 * softmax[l] * sig_norm ; split hi/lo, scatter to WMMA order
  {
    int l = t >> 3;
    int f0 = (t & 7) * 8;
    float x[8]; float ssq = 0.f;
    for (int i = 0; i < 8; ++i) {
      float v = valid ? sig[(c * Ln + l) * Fn + f0 + i] : 0.0f;
      x[i] = v; ssq += v * v;
    }
    for (int o = 1; o < 8; o <<= 1) ssq += __shfl_xor(ssq, o, 32);
    float scale = valid ? (0.5f * s_p[l] / fmaxf(sqrtf(ssq), 1e-12f)) : 0.0f;
    for (int i = 0; i < 8; ++i) {
      float w = x[i] * scale;
      unsigned short hi = f2bf(w);
      float lo = w - bf2f(hi);
      int idx = stagedB_idx(c, l * Fn + f0 + i, KS);
      Bhi[idx] = hi;
      Blo[idx] = f2bf(lo);
    }
  }

  // hier targets: 0.25 * l2norm(mean over half levels)
  if (t < 2 * Fn) {
    int f = t & (Fn - 1); int half = t >> 6;
    float s = 0.f;
    if (valid)
      for (int l = 0; l < 16; ++l) s += sig[(c * Ln + half * 16 + l) * Fn + f];
    s_vec[t] = s * (1.0f / 16.0f);
  }
  __syncthreads();
  if (t < 2) {
    float q = 0.f;
    for (int f = 0; f < Fn; ++f) { float v = s_vec[t * Fn + f]; q += v * v; }
    s_ssq[t] = q;
  }
  __syncthreads();
  if (t < 2 * Fn) {
    int half = t >> 6;
    float scale = valid ? (0.25f / fmaxf(sqrtf(s_ssq[half]), 1e-12f)) : 0.0f;
    float w = s_vec[t] * scale;
    unsigned short hi = f2bf(w);
    float lo = w - bf2f(hi);
    int idx = stagedB_idx(c, t, KSH);     // kk = half*64 + f == t
    BHhi[idx] = hi;
    BHlo[idx] = f2bf(lo);
  }
  if (t < Ln - 1) {                       // expected[c,l] = sigmoid(r_l - r_{l+1})
    float v = valid ? (1.0f / (1.0f + __expf(-(s_raw[t] - s_raw[t + 1])))) : 0.0f;
    exT[t * Cp + c] = v;
  }
}

// =====================  batch-side preprocessing  =====================
__global__ void __launch_bounds__(256) prep_rows(
    const float* __restrict__ pe,
    unsigned short* __restrict__ Ahi, unsigned short* __restrict__ Alo,
    unsigned short* __restrict__ AHhi, unsigned short* __restrict__ AHlo,
    float* __restrict__ ssT)
{
  int b = blockIdx.x;
  int t = threadIdx.x;
  __shared__ float s_x[Ln * Fn];
  __shared__ float s_nrm[Ln];
  __shared__ float s_vec[2 * Fn];
  __shared__ float s_ssq[2];

  int l = t >> 3;
  int f0 = (t & 7) * 8;
  float x[8]; float ssq = 0.f;
  for (int i = 0; i < 8; ++i) {
    float v = pe[(b * Ln + l) * Fn + f0 + i];
    x[i] = v; s_x[l * Fn + f0 + i] = v; ssq += v * v;
  }
  for (int o = 1; o < 8; o <<= 1) ssq += __shfl_xor(ssq, o, 32);
  float nrm = sqrtf(ssq);
  if ((t & 7) == 0) s_nrm[l] = nrm;
  float scale = 1.0f / fmaxf(nrm, 1e-12f);
  for (int i = 0; i < 8; ++i) {
    float w = x[i] * scale;
    unsigned short hi = f2bf(w);
    float lo = w - bf2f(hi);
    int idx = stagedA_idx(b, l * Fn + f0 + i, KS);
    Ahi[idx] = hi;
    Alo[idx] = f2bf(lo);
  }
  __syncthreads();

  // self_sim (adjacent-level cosine), stored transposed [l][B]
  if (l < Ln - 1) {
    float d = 0.f;
    for (int i = 0; i < 8; ++i) d += x[i] * s_x[(l + 1) * Fn + f0 + i];
    for (int o = 1; o < 8; o <<= 1) d += __shfl_xor(d, o, 32);
    if ((t & 7) == 0) {
      float na = fmaxf(s_nrm[l], 1e-8f);
      float nb = fmaxf(s_nrm[l + 1], 1e-8f);
      ssT[l * Bn + b] = (d / (na * nb) + 1.0f) * 0.5f;
    }
  }

  // hier query: l2norm(mean over half levels), no extra scale (0.25 on B side)
  if (t < 2 * Fn) {
    int f = t & (Fn - 1); int half = t >> 6;
    float s = 0.f;
    for (int ll = 0; ll < 16; ++ll) s += s_x[(half * 16 + ll) * Fn + f];
    s_vec[t] = s * (1.0f / 16.0f);
  }
  __syncthreads();
  if (t < 2) {
    float q = 0.f;
    for (int f = 0; f < Fn; ++f) { float v = s_vec[t * Fn + f]; q += v * v; }
    s_ssq[t] = q;
  }
  __syncthreads();
  if (t < 2 * Fn) {
    int half = t >> 6;
    float sc = 1.0f / fmaxf(sqrtf(s_ssq[half]), 1e-12f);
    float w = s_vec[t] * sc;
    unsigned short hi = f2bf(w);
    float lo = w - bf2f(hi);
    int idx = stagedA_idx(b, t, KSH);
    AHhi[idx] = hi;
    AHlo[idx] = f2bf(lo);
  }
}

// =====================  fused GEMM + epilogue  =====================
// block = 8 waves; block tile = 128 rows (B) x 32 cols (C); wave tile = 16x32.
__global__ void __launch_bounds__(256) gemm_fuse(
    const unsigned short* __restrict__ Ahi, const unsigned short* __restrict__ Alo,
    const unsigned short* __restrict__ Bh , const unsigned short* __restrict__ Bl ,
    const unsigned short* __restrict__ AHh, const unsigned short* __restrict__ AHl,
    const unsigned short* __restrict__ BHh, const unsigned short* __restrict__ BHl,
    const float* __restrict__ ssT, const float* __restrict__ exT,
    const float* __restrict__ cm, const float* __restrict__ protos,
    float* __restrict__ out)
{
  __shared__ __align__(16) float s_ss[31 * 128];  // [l][row_local]
  __shared__ float s_ex[31 * 32];                 // [l][col_local]
  __shared__ float s_cm[128];
  __shared__ float s_pr[32];

  int tid  = threadIdx.x;
  int w    = tid >> 5;
  int lane = tid & 31;
  int b0   = blockIdx.y * 128;
  int c0   = blockIdx.x * 32;

  for (int i = tid; i < 31 * 128; i += 256) {
    int l = i >> 7, mb = i & 127;
    s_ss[i] = ssT[l * Bn + b0 + mb];
  }
  for (int i = tid; i < 31 * 32; i += 256) {
    int l = i >> 5, n = i & 31;
    s_ex[i] = exT[l * Cp + c0 + n];
  }
  if (tid < 128) s_cm[tid] = cm[b0 + tid];
  if (tid < 32)  s_pr[tid] = (c0 + tid < Cn) ? protos[c0 + tid] : 0.0f;
  __syncthreads();

  int bt  = (b0 >> 4) + w;                 // this wave's B row-tile
  int ct0 = (c0 >> 4);                     // two C col-tiles: ct0, ct0+1

  const v16bf* aH  = (const v16bf*)Ahi + (size_t)bt  * KS  * 32 + lane;
  const v16bf* aL  = (const v16bf*)Alo + (size_t)bt  * KS  * 32 + lane;
  const v16bf* q0H = (const v16bf*)Bh  + (size_t)ct0 * KS  * 32 + lane;
  const v16bf* q0L = (const v16bf*)Bl  + (size_t)ct0 * KS  * 32 + lane;
  const v16bf* q1H = q0H + KS * 32;
  const v16bf* q1L = q0L + KS * 32;

  v8f acc0 = {}, acc1 = {}, hac0 = {}, hac1 = {};

  for (int ks4 = 0; ks4 < KS; ks4 += 4) {
    // Stream A 8..11 k-steps ahead into the WGP cache. All four slots are
    // immediate offsets off the same base VGPR pair the body's first loads
    // use, so no extra address VALU and no WMMA WAR hazard NOPs. The tail
    // runs past the tile but stays inside the contiguous workspace, and
    // prefetch is speculative => always safe.
    const v16bf* pa = &aH[ks4 * 32];
    const v16bf* pl = &aL[ks4 * 32];
    PF(pa, 8192); PF(pa, 9216); PF(pa, 10240); PF(pa, 11264);
    PF(pl, 8192); PF(pl, 9216); PF(pl, 10240); PF(pl, 11264);
    #pragma unroll
    for (int u = 0; u < 4; ++u) {
      int ks = ks4 + u;
      v16bf ah  = aH [ks * 32];
      v16bf al  = aL [ks * 32];
      v16bf bh0 = q0H[ks * 32];
      v16bf bl0 = q0L[ks * 32];
      v16bf bh1 = q1H[ks * 32];
      v16bf bl1 = q1L[ks * 32];
      // hi*hi + hi*lo + lo*hi  (lo*lo term negligible: ~2^-16 relative)
      acc0 = __builtin_amdgcn_wmma_f32_16x16x32_bf16(false, ah, false, bh0, (short)0, acc0, false, false);
      acc0 = __builtin_amdgcn_wmma_f32_16x16x32_bf16(false, ah, false, bl0, (short)0, acc0, false, false);
      acc0 = __builtin_amdgcn_wmma_f32_16x16x32_bf16(false, al, false, bh0, (short)0, acc0, false, false);
      acc1 = __builtin_amdgcn_wmma_f32_16x16x32_bf16(false, ah, false, bh1, (short)0, acc1, false, false);
      acc1 = __builtin_amdgcn_wmma_f32_16x16x32_bf16(false, ah, false, bl1, (short)0, acc1, false, false);
      acc1 = __builtin_amdgcn_wmma_f32_16x16x32_bf16(false, al, false, bh1, (short)0, acc1, false, false);
    }
  }

  const v16bf* hAh = (const v16bf*)AHh + (size_t)bt  * KSH * 32 + lane;
  const v16bf* hAl = (const v16bf*)AHl + (size_t)bt  * KSH * 32 + lane;
  const v16bf* hBh = (const v16bf*)BHh + (size_t)ct0 * KSH * 32 + lane;
  const v16bf* hBl = (const v16bf*)BHl + (size_t)ct0 * KSH * 32 + lane;
  #pragma unroll
  for (int ks = 0; ks < KSH; ++ks) {
    v16bf ah  = hAh[ks * 32];
    v16bf al  = hAl[ks * 32];
    v16bf bh0 = hBh[ks * 32];
    v16bf bl0 = hBl[ks * 32];
    v16bf bh1 = hBh[(KSH + ks) * 32];
    v16bf bl1 = hBl[(KSH + ks) * 32];
    hac0 = __builtin_amdgcn_wmma_f32_16x16x32_bf16(false, ah, false, bh0, (short)0, hac0, false, false);
    hac0 = __builtin_amdgcn_wmma_f32_16x16x32_bf16(false, ah, false, bl0, (short)0, hac0, false, false);
    hac0 = __builtin_amdgcn_wmma_f32_16x16x32_bf16(false, al, false, bh0, (short)0, hac0, false, false);
    hac1 = __builtin_amdgcn_wmma_f32_16x16x32_bf16(false, ah, false, bh1, (short)0, hac1, false, false);
    hac1 = __builtin_amdgcn_wmma_f32_16x16x32_bf16(false, ah, false, bl1, (short)0, hac1, false, false);
    hac1 = __builtin_amdgcn_wmma_f32_16x16x32_bf16(false, al, false, bh1, (short)0, hac1, false, false);
  }

  // --------- epilogue ---------
  int hh = lane >> 4, n = lane & 15;
  int mbase = w * 16 + 8 * hh;             // LDS row base for this lane's 8 rows
  float sd0[8], sd1[8];
  for (int r = 0; r < 8; ++r) { sd0[r] = 0.f; sd1[r] = 0.f; }
  for (int l = 0; l < 31; ++l) {
    float e0 = s_ex[l * 32 + n];
    float e1 = s_ex[l * 32 + 16 + n];
    const float4* pv = (const float4*)&s_ss[l * 128 + mbase];
    float4 v0 = pv[0], v1 = pv[1];
    float vs[8] = { v0.x, v0.y, v0.z, v0.w, v1.x, v1.y, v1.z, v1.w };
    for (int r = 0; r < 8; ++r) {
      sd0[r] += fabsf(vs[r] - e0);
      sd1[r] += fabsf(vs[r] - e1);
    }
  }

  const float eps = 1e-6f;
  int cg0 = c0 + n, cg1 = c0 + 16 + n;
  float pr0 = s_pr[n], pr1 = s_pr[16 + n];
  for (int r = 0; r < 8; ++r) {
    int mloc = w * 16 + 8 * hh + r;
    int bg = b0 + mloc;
    float cmv = s_cm[mloc];
    {
      float tv = fminf(fmaxf(acc0[r] + 0.5f, eps), 1.0f);
      float sv = fminf(fmaxf(1.0f - sd0[r] * (1.0f / 31.0f), eps), 1.0f);
      float dd = cmv - pr0;
      float cv = fminf(fmaxf(__expf(-dd * dd * 10.0f) + 1e-8f, eps), 1.0f);
      float hv = fminf(fmaxf(hac0[r] * 0.25f + 0.5f, eps), 1.0f);
      if (cg0 < Cn) out[(size_t)bg * Cn + cg0] = sqrtf(sqrtf(tv * sv * cv * hv));
    }
    {
      float tv = fminf(fmaxf(acc1[r] + 0.5f, eps), 1.0f);
      float sv = fminf(fmaxf(1.0f - sd1[r] * (1.0f / 31.0f), eps), 1.0f);
      float dd = cmv - pr1;
      float cv = fminf(fmaxf(__expf(-dd * dd * 10.0f) + 1e-8f, eps), 1.0f);
      float hv = fminf(fmaxf(hac1[r] * 0.25f + 0.5f, eps), 1.0f);
      if (cg1 < Cn) out[(size_t)bg * Cn + cg1] = sqrtf(sqrtf(tv * sv * cv * hv));
    }
  }
}

extern "C" void kernel_launch(void* const* d_in, const int* in_sizes, int n_in,
                              void* d_out, int out_size, void* d_ws, size_t ws_size,
                              hipStream_t stream) {
  (void)in_sizes; (void)n_in; (void)out_size; (void)ws_size;
  const float* pe     = (const float*)d_in[0];   // (B,L,F)
  const float* cm     = (const float*)d_in[1];   // (B,)
  const float* sig    = (const float*)d_in[2];   // (C,L,F)
  const float* protos = (const float*)d_in[3];   // (C,)
  const float* lres   = (const float*)d_in[4];   // (C,L)
  float* out = (float*)d_out;

  char* ws = (char*)d_ws;
  unsigned short* Ahi  = (unsigned short*)(ws + OFF_AHI);
  unsigned short* Alo  = (unsigned short*)(ws + OFF_ALO);
  unsigned short* Bhi  = (unsigned short*)(ws + OFF_BHI);
  unsigned short* Blo  = (unsigned short*)(ws + OFF_BLO);
  unsigned short* AHhi = (unsigned short*)(ws + OFF_AHHI);
  unsigned short* AHlo = (unsigned short*)(ws + OFF_AHLO);
  unsigned short* BHhi = (unsigned short*)(ws + OFF_BHHI);
  unsigned short* BHlo = (unsigned short*)(ws + OFF_BHLO);
  float* ssT = (float*)(ws + OFF_SST);
  float* exT = (float*)(ws + OFF_EXT);

  hipLaunchKernelGGL(prep_classes, dim3(Cp), dim3(256), 0, stream,
                     sig, lres, Bhi, Blo, BHhi, BHlo, exT);
  hipLaunchKernelGGL(prep_rows, dim3(Bn), dim3(256), 0, stream,
                     pe, Ahi, Alo, AHhi, AHlo, ssT);
  hipLaunchKernelGGL(gemm_fuse, dim3(Cp / 32, Bn / 128), dim3(256), 0, stream,
                     Ahi, Alo, Bhi, Blo, AHhi, AHlo, BHhi, BHlo,
                     ssT, exT, cm, protos, out);
}
